// BasicAttetnionLayer_50113678410479
// MI455X (gfx1250) — matrix operand order repaired and checked by reading, not compiled, not run on hardware
//
#include <hip/hip_runtime.h>
#include <hip/hip_bf16.h>

// ---------------------------------------------------------------------------
// Types for WMMA fragments (gfx1250, wave32)
// ---------------------------------------------------------------------------
typedef _Float16 half8  __attribute__((ext_vector_type(8)));
typedef _Float16 v16h   __attribute__((ext_vector_type(16)));
typedef float    v8f    __attribute__((ext_vector_type(8)));

#define CCH   128
#define NHEAD 8
#define HD    16
#define FFN   512
#define BSZ   4
#define H2    96
#define W2    96
#define P2    (H2*W2)          // 9216 kv tokens per batch
#define HQ    16
#define WQ    16
#define PQ    (HQ*WQ)          // 256 q tokens per batch
#define GN_GROUPS 8
#define EPSV  1e-5f

__device__ inline v8f wmma_f16(v16h a, v16h b, v8f c) {
  // v_wmma_f32_16x16x32_f16 : D = A(16x32 f16) * B(32x16 f16) + C(16x16 f32)
  return __builtin_amdgcn_wmma_f32_16x16x32_f16(
      /*neg_a=*/false, a, /*neg_b=*/false, b,
      /*c_mod=*/(short)0, c, /*reuse_a=*/false, /*reuse_b=*/false);
}

// PositionEmbeddingSine (normalize=False, temperature=10000), channel c in [0,128)
__device__ inline float sine_pos_val(int h, int w, int c) {
  float p  = (c < 64) ? (float)(h + 1) : (float)(w + 1);
  int   cc = c & 63;
  int   j  = cc >> 1;
  float dimt = powf(10000.0f, (float)j / 32.0f);   // 10000^(2*floor(c/2)/64)
  float arg  = p / dimt;
  return (cc & 1) ? cosf(arg) : sinf(arg);
}

// ---------------------------------------------------------------------------
// Weight convert f32 -> f16 (kept in [N_out, K_in] row-major: B-fragment
// (32x16) lane = output column, 16 contiguous K halves = one 32B load)
// ---------------------------------------------------------------------------
__global__ void cvt_f32_f16_kernel(const float* __restrict__ src,
                                   _Float16* __restrict__ dst, int n) {
  int i = blockIdx.x * blockDim.x + threadIdx.x;
  if (i < n) dst[i] = (_Float16)src[i];
}

// ---------------------------------------------------------------------------
// Token prep for the 96x96 grid:
//   up     = bilinear(x1, 96x96) (align_corners)
//   Acat   : f16 [B*9216, 256]  = {up, x2}   (concat conv input)
//   x2tok  : f32 [B*9216, 128]  (residual for concat conv)
//   kvtok  : f16 [B*9216, 128]  = x2 + sine_pos (k/v projection input)
// ---------------------------------------------------------------------------
__global__ void prep_tokens_kernel(const float* __restrict__ x1,
                                   const float* __restrict__ x2,
                                   _Float16* __restrict__ Acat,
                                   float* __restrict__ x2tok,
                                   _Float16* __restrict__ kvtok) {
  int idx = blockIdx.x * blockDim.x + threadIdx.x;     // B*9216*128
  if (idx >= BSZ * P2 * CCH) return;
  int c = idx & (CCH - 1);
  int p = (idx >> 7) % P2;
  int b = idx / (P2 * CCH);
  int hh = p / W2, ww = p % W2;

  // bilinear 64 -> 96, align_corners=True
  const float ratio = 63.0f / 95.0f;
  float fy = hh * ratio, fx = ww * ratio;
  int y0 = (int)floorf(fy), x0 = (int)floorf(fx);
  int y1 = min(y0 + 1, 63), x1i = min(x0 + 1, 63);
  float wy = fy - (float)y0, wx = fx - (float)x0;
  const float* xc = x1 + ((size_t)(b * CCH + c)) * 64 * 64;
  float v00 = xc[y0 * 64 + x0],  v01 = xc[y0 * 64 + x1i];
  float v10 = xc[y1 * 64 + x0],  v11 = xc[y1 * 64 + x1i];
  float up = (v00 * (1.f - wy) + v10 * wy) * (1.f - wx)
           + (v01 * (1.f - wy) + v11 * wy) * wx;

  float x2v = x2[((size_t)(b * CCH + c)) * P2 + p];
  size_t t = (size_t)b * P2 + p;
  Acat[t * (2 * CCH) + c]        = (_Float16)up;
  Acat[t * (2 * CCH) + CCH + c]  = (_Float16)x2v;
  x2tok[t * CCH + c]             = x2v;
  kvtok[t * CCH + c]             = (_Float16)(x2v + sine_pos_val(hh, ww, c));
}

// ---------------------------------------------------------------------------
// Q-token prep: 4x4 avg-pool of x1 -> x1s (f32 master), qtok = pooled + pos
// ---------------------------------------------------------------------------
__global__ void prep_q_kernel(const float* __restrict__ x1,
                              float* __restrict__ x1s,
                              _Float16* __restrict__ qtok) {
  int idx = blockIdx.x * blockDim.x + threadIdx.x;     // B*256*128
  if (idx >= BSZ * PQ * CCH) return;
  int c = idx & (CCH - 1);
  int t = (idx >> 7) & (PQ - 1);
  int b = idx >> 15;
  int ty = t / WQ, tx = t % WQ;
  const float* xc = x1 + (size_t)(b * CCH + c) * 64 * 64;
  float s = 0.f;
#pragma unroll
  for (int dy = 0; dy < 4; ++dy)
#pragma unroll
    for (int dx = 0; dx < 4; ++dx)
      s += xc[(ty * 4 + dy) * 64 + (tx * 4 + dx)];
  s *= (1.0f / 16.0f);
  size_t o = (size_t)(b * PQ + t) * CCH + c;
  x1s[o]  = s;
  qtok[o] = (_Float16)(s + sine_pos_val(ty, tx, c));
}

// ---------------------------------------------------------------------------
// Strip WMMA GEMM: out[M,N] = A[M,K](f16 row-major) * Bw[N,K]^T (+bias,
// +resid, optional relu).  Each wave owns a 16-row x 128-col strip:
// one A fragment per k-step feeds 8 back-to-back v_wmma against 8 weight
// fragments (8x A-traffic reuse; weights are tiny and L2-resident).
// Requires N % 128 == 0 (true for all GEMMs in this model).
// ---------------------------------------------------------------------------
__global__ __launch_bounds__(256)
void gemm_wmma_kernel(const _Float16* __restrict__ A,
                      const _Float16* __restrict__ Bw,   // [N, K] row-major
                      const float* __restrict__ bias,
                      const float* __restrict__ resid,   // [M, N] f32 or null
                      float* __restrict__ outF,          // [M, N] f32 or null
                      _Float16* __restrict__ outH,       // [M, N] f16 or null
                      int M, int N, int K, int relu) {
  int wave = (int)((blockIdx.x * blockDim.x + threadIdx.x) >> 5);
  int lane = threadIdx.x & 31;
  int nstrips = N >> 7;                       // 128-column strips
  int total = (M >> 4) * nstrips;
  if (wave >= total) return;
  int tm = wave / nstrips, ts = wave - tm * nstrips;
  int m0 = tm << 4, n0 = ts << 7;
  int row = lane & 15, kh = lane >> 4;

  const _Float16* arow  = A  + (size_t)(m0 + row) * K;   // A frag: lane row
  const _Float16* bbase = Bw + (size_t)(n0 + row) * K;   // B frag: lane col

  v8f acc[8];
#pragma unroll
  for (int i = 0; i < 8; ++i) acc[i] = (v8f){};

  for (int k0 = 0; k0 < K; k0 += 32) {
    // A 16x32: elem e -> K = kh*16+e (contiguous, matches B)
    half8 a0 = *(const half8*)(arow + k0 + kh * 16);
    half8 a1 = *(const half8*)(arow + k0 + kh * 16 + 8);
    v16h af;
#pragma unroll
    for (int i = 0; i < 8; ++i) { af[i] = a0[i]; af[8 + i] = a1[i]; }
#pragma unroll
    for (int nt = 0; nt < 8; ++nt) {
      // B 32x16: elem e -> K = kh*16+e (16 contiguous halves = 32B load)
      v16h bf = *(const v16h*)(bbase + (size_t)nt * 16 * K + k0 + kh * 16);
      acc[nt] = wmma_f16(af, bf, acc[nt]);
    }
  }

  int col = lane & 15;
#pragma unroll
  for (int nt = 0; nt < 8; ++nt) {
#pragma unroll
    for (int r = 0; r < 8; ++r) {
      int rr = r + 8 * kh;          // C layout: VGPR r holds rows r / r+8
      size_t idx = (size_t)(m0 + rr) * N + n0 + nt * 16 + col;
      float v = acc[nt][r];
      if (bias)  v += bias[n0 + nt * 16 + col];
      if (resid) v += resid[idx];
      if (relu)  v = fmaxf(v, 0.0f);
      if (outF) outF[idx] = v;
      if (outH) outH[idx] = (_Float16)v;
    }
  }
}

// ---------------------------------------------------------------------------
// GroupNorm: stats per (b, group) then normalize
// ---------------------------------------------------------------------------
__global__ void gn_stats_kernel(const float* __restrict__ pre,   // [B*9216,128]
                                float* __restrict__ stats) {     // [32][2]
  int bg = blockIdx.x;           // b*8 + g
  int b = bg >> 3, g = bg & 7;
  __shared__ float ss[256], sq[256];
  float s = 0.f, q = 0.f;
  for (int ch = 0; ch < 16; ++ch) {
    int cc = g * 16 + ch;
    const float* base = pre + (size_t)b * P2 * CCH + cc;
    for (int p = threadIdx.x; p < P2; p += 256) {
      float v = base[(size_t)p * CCH];
      s += v; q += v * v;
    }
  }
  ss[threadIdx.x] = s; sq[threadIdx.x] = q;
  __syncthreads();
  for (int st = 128; st > 0; st >>= 1) {
    if ((int)threadIdx.x < st) { ss[threadIdx.x] += ss[threadIdx.x + st];
                                 sq[threadIdx.x] += sq[threadIdx.x + st]; }
    __syncthreads();
  }
  if (threadIdx.x == 0) {
    float n = 16.0f * (float)P2;
    float m = ss[0] / n;
    stats[bg * 2 + 0] = m;
    stats[bg * 2 + 1] = sq[0] / n - m * m;
  }
}

__global__ void gn_norm_kernel(const float* __restrict__ pre,
                               const float* __restrict__ stats,
                               const float* __restrict__ gw,
                               const float* __restrict__ gb,
                               float* __restrict__ shortcut) {
  int idx = blockIdx.x * blockDim.x + threadIdx.x;   // B*9216*128
  if (idx >= BSZ * P2 * CCH) return;
  int c = idx & (CCH - 1);
  int b = idx / (P2 * CCH);
  int g = c >> 4;
  float m = stats[(b * 8 + g) * 2 + 0];
  float v = stats[(b * 8 + g) * 2 + 1];
  shortcut[idx] = (pre[idx] - m) * rsqrtf(v + EPSV) * gw[c] + gb[c];
}

// ---------------------------------------------------------------------------
// Flash attention: 1 wave per (b, head, 16-row q tile), stream 32 kv / iter.
// S = Q*K^T (no 1/sqrt(d) scaling, as in reference), online softmax, O += P*V.
// Accumulates O into x1s (which holds the pooled x1).
// ---------------------------------------------------------------------------
__global__ __launch_bounds__(32)
void attention_kernel(const _Float16* __restrict__ Q,   // [B*256, 128]
                      const _Float16* __restrict__ Km,  // [B*9216,128]
                      const _Float16* __restrict__ Vm,  // [B*9216,128]
                      float* __restrict__ x1s) {        // [B*256, 128] +=
  __shared__ float    sS[16][32];
  __shared__ _Float16 sP[16][32];
  __shared__ float    sM[16], sL[16], sScale[16];

  int bid   = blockIdx.x;
  int qtile = bid & 15;
  int h     = (bid >> 4) & 7;
  int b     = bid >> 7;
  int lane  = threadIdx.x;
  int row   = lane & 15, kh = lane >> 4;

  // Q fragment (K-dim = head dim 16, zero-padded to 32)
  const _Float16* qptr =
      Q + (size_t)(b * PQ + qtile * 16 + row) * CCH + h * HD;
  v16h aq;
#pragma unroll
  for (int i = 0; i < 16; ++i) aq[i] = (_Float16)0;
  if (kh == 0) {
    half8 q8 = *(const half8*)(qptr);
    half8 q9 = *(const half8*)(qptr + 8);
#pragma unroll
    for (int i = 0; i < 8; ++i) { aq[i] = q8[i]; aq[8 + i] = q9[i]; }
  }
  if (lane < 16) { sM[lane] = -1e30f; sL[lane] = 0.0f; }
  __syncthreads();

  v8f o = {};
  for (int t0 = 0; t0 < P2; t0 += 32) {
    // ---- S tiles: cols 0..15 and 16..31 -------------------------------
    v16h bk0, bk1;
#pragma unroll
    for (int i = 0; i < 16; ++i) { bk0[i] = (_Float16)0; bk1[i] = (_Float16)0; }
    if (kh == 0) {   // only K-dims 0..15 are nonzero (head dim = 16)
      bk0 = *(const v16h*)(Km + (size_t)(b * P2 + t0 +      row) * CCH + h * HD);
      bk1 = *(const v16h*)(Km + (size_t)(b * P2 + t0 + 16 + row) * CCH + h * HD);
    }
    v8f s0 = {}, s1 = {};
    s0 = wmma_f16(aq, bk0, s0);
    s1 = wmma_f16(aq, bk1, s1);

#pragma unroll
    for (int r = 0; r < 8; ++r) {
      sS[r + 8 * kh][row]      = s0[r];
      sS[r + 8 * kh][16 + row] = s1[r];
    }
    __syncthreads();

    // ---- online softmax (lanes 0..15, one row each) -------------------
    if (lane < 16) {
      float mold = sM[lane];
      float nm = mold;
#pragma unroll
      for (int j = 0; j < 32; ++j) nm = fmaxf(nm, sS[lane][j]);
      float scale = expf(mold - nm);
      float sum = 0.f;
#pragma unroll
      for (int j = 0; j < 32; ++j) {
        float p = expf(sS[lane][j] - nm);
        sP[lane][j] = (_Float16)p;
        sum += p;
      }
      sL[lane] = sL[lane] * scale + sum;
      sScale[lane] = scale;
      sM[lane] = nm;
    }
    __syncthreads();

    // rescale running O
#pragma unroll
    for (int r = 0; r < 8; ++r) o[r] *= sScale[r + 8 * kh];

    // ---- P fragment (A layout 16x32) ----------------------------------
    v16h ap;
#pragma unroll
    for (int i = 0; i < 8; ++i) {
      ap[i]     = sP[row][kh * 16 + i];
      ap[8 + i] = sP[row][kh * 16 + 8 + i];
    }
    // ---- V fragment (B layout 32x16; col = head dim) ------------------
    v16h bv;
#pragma unroll
    for (int e = 0; e < 16; ++e)
      bv[e] = Vm[(size_t)(b * P2 + t0 + kh * 16 + e) * CCH + h * HD + row];

    o = wmma_f16(ap, bv, o);
    __syncthreads();
  }

  // ---- finalize: x1s += O / l ------------------------------------------
#pragma unroll
  for (int r = 0; r < 8; ++r) {
    int rr = r + 8 * kh;
    float val = o[r] / sL[rr];
    size_t idx = (size_t)(b * PQ + qtile * 16 + rr) * CCH + h * HD + (lane & 15);
    x1s[idx] += val;
  }
}

// ---------------------------------------------------------------------------
// Bilinear 16x16 -> 96x96 from token-major f32 into token-major f16
// ---------------------------------------------------------------------------
__global__ void upsample2_kernel(const float* __restrict__ x1s2,   // [B*256,128]
                                 _Float16* __restrict__ up2) {     // [B*9216,128]
  int idx = blockIdx.x * blockDim.x + threadIdx.x;   // B*9216*128
  if (idx >= BSZ * P2 * CCH) return;
  int c = idx & (CCH - 1);
  int p = (idx >> 7) % P2;
  int b = idx / (P2 * CCH);
  int hh = p / W2, ww = p % W2;
  const float ratio = 15.0f / 95.0f;
  float fy = hh * ratio, fx = ww * ratio;
  int y0 = (int)floorf(fy), x0 = (int)floorf(fx);
  int y1 = min(y0 + 1, 15), x1i = min(x0 + 1, 15);
  float wy = fy - (float)y0, wx = fx - (float)x0;
  const float* base = x1s2 + (size_t)b * PQ * CCH + c;
  float v00 = base[(size_t)(y0 * WQ + x0)  * CCH];
  float v01 = base[(size_t)(y0 * WQ + x1i) * CCH];
  float v10 = base[(size_t)(y1 * WQ + x0)  * CCH];
  float v11 = base[(size_t)(y1 * WQ + x1i) * CCH];
  float up = (v00 * (1.f - wy) + v10 * wy) * (1.f - wx)
           + (v01 * (1.f - wy) + v11 * wy) * wx;
  up2[idx] = (_Float16)up;
}

// Token-major [B*9216, 128] -> BCHW output
__global__ void permute_out_kernel(const float* __restrict__ ftok,
                                   float* __restrict__ out) {
  int idx = blockIdx.x * blockDim.x + threadIdx.x;   // B*128*9216
  if (idx >= BSZ * CCH * P2) return;
  int p = idx % P2;
  int c = (idx / P2) & (CCH - 1);
  int b = idx / (CCH * P2);
  out[idx] = ftok[(size_t)(b * P2 + p) * CCH + c];
}

// ---------------------------------------------------------------------------
// Host driver
// ---------------------------------------------------------------------------
extern "C" void kernel_launch(void* const* d_in, const int* in_sizes, int n_in,
                              void* d_out, int out_size, void* d_ws, size_t ws_size,
                              hipStream_t stream) {
  const float* x1       = (const float*)d_in[0];
  const float* x2       = (const float*)d_in[1];
  const float* concat_w = (const float*)d_in[2];
  const float* concat_b = (const float*)d_in[3];
  const float* gn_w     = (const float*)d_in[4];
  const float* gn_b     = (const float*)d_in[5];
  const float* q_w      = (const float*)d_in[6];
  const float* q_b      = (const float*)d_in[7];
  const float* k_w      = (const float*)d_in[8];
  const float* k_b      = (const float*)d_in[9];
  const float* v_w      = (const float*)d_in[10];
  const float* v_b      = (const float*)d_in[11];
  const float* fc1_w    = (const float*)d_in[12];
  const float* fc1_b    = (const float*)d_in[13];
  const float* fc2_w    = (const float*)d_in[14];
  const float* fc2_b    = (const float*)d_in[15];
  const float* out_w    = (const float*)d_in[16];
  const float* out_b    = (const float*)d_in[17];
  float* out = (float*)d_out;

  char* ws = (char*)d_ws;
  size_t off = 0;
  auto alloc = [&](size_t bytes) {
    size_t r = off;
    off = (off + bytes + 255) & ~(size_t)255;
    return r;
  };
  const size_t NT2 = (size_t)BSZ * P2;   // 36864 kv tokens
  const size_t NTQ = (size_t)BSZ * PQ;   // 1024 q tokens

  _Float16* wConcat16 = (_Float16*)(ws + alloc(CCH * 2 * CCH * 2));
  _Float16* wQ16      = (_Float16*)(ws + alloc(CCH * CCH * 2));
  _Float16* wK16      = (_Float16*)(ws + alloc(CCH * CCH * 2));
  _Float16* wV16      = (_Float16*)(ws + alloc(CCH * CCH * 2));
  _Float16* wFc1_16   = (_Float16*)(ws + alloc(FFN * CCH * 2));
  _Float16* wFc2_16   = (_Float16*)(ws + alloc(CCH * FFN * 2));
  _Float16* wOut16    = (_Float16*)(ws + alloc(CCH * CCH * 2));
  _Float16* Acat      = (_Float16*)(ws + alloc(NT2 * 2 * CCH * 2));
  float*    x2tok     = (float*)   (ws + alloc(NT2 * CCH * 4));
  _Float16* kvtok     = (_Float16*)(ws + alloc(NT2 * CCH * 2));
  _Float16* qtok      = (_Float16*)(ws + alloc(NTQ * CCH * 2));
  float*    pre       = (float*)   (ws + alloc(NT2 * CCH * 4));
  float*    stats     = (float*)   (ws + alloc(BSZ * GN_GROUPS * 2 * 4));
  float*    shortcut  = (float*)   (ws + alloc(NT2 * CCH * 4));
  _Float16* qproj     = (_Float16*)(ws + alloc(NTQ * CCH * 2));
  _Float16* kproj     = (_Float16*)(ws + alloc(NT2 * CCH * 2));
  _Float16* vproj     = (_Float16*)(ws + alloc(NT2 * CCH * 2));
  float*    x1s       = (float*)   (ws + alloc(NTQ * CCH * 4));
  _Float16* x1s16     = (_Float16*)(ws + alloc(NTQ * CCH * 2));
  _Float16* hidden    = (_Float16*)(ws + alloc(NTQ * FFN * 2));
  float*    x1s2      = (float*)   (ws + alloc(NTQ * CCH * 4));
  _Float16* up2       = (_Float16*)(ws + alloc(NT2 * CCH * 2));
  float*    ftok      = (float*)   (ws + alloc(NT2 * CCH * 4));
  (void)ws_size; (void)n_in; (void)in_sizes; (void)out_size;

  auto cvt = [&](const float* s, _Float16* d, int n) {
    cvt_f32_f16_kernel<<<(n + 255) / 256, 256, 0, stream>>>(s, d, n);
  };
  cvt(concat_w, wConcat16, CCH * 2 * CCH);
  cvt(q_w,  wQ16,  CCH * CCH);
  cvt(k_w,  wK16,  CCH * CCH);
  cvt(v_w,  wV16,  CCH * CCH);
  cvt(fc1_w, wFc1_16, FFN * CCH);
  cvt(fc2_w, wFc2_16, CCH * FFN);
  cvt(out_w, wOut16, CCH * CCH);

  // token prep
  {
    int n = BSZ * P2 * CCH;
    prep_tokens_kernel<<<(n + 255) / 256, 256, 0, stream>>>(x1, x2, Acat, x2tok, kvtok);
  }
  {
    int n = BSZ * PQ * CCH;
    prep_q_kernel<<<(n + 255) / 256, 256, 0, stream>>>(x1, x1s, qtok);
  }

  auto gemm = [&](const _Float16* A, const _Float16* Bw, const float* bias,
                  const float* resid, float* oF, _Float16* oH,
                  int M, int N, int K, int relu) {
    int strips = (M / 16) * (N / 128);     // one 16x128 strip per wave
    int blocks = (strips + 7) / 8;         // 8 waves / 256-thread block
    gemm_wmma_kernel<<<blocks, 256, 0, stream>>>(A, Bw, bias, resid, oF, oH,
                                                 M, N, K, relu);
  };

  // concat conv (+x2 residual) -> pre; GroupNorm -> shortcut
  gemm(Acat, wConcat16, concat_b, x2tok, pre, nullptr,
       (int)NT2, CCH, 2 * CCH, 0);
  gn_stats_kernel<<<BSZ * GN_GROUPS, 256, 0, stream>>>(pre, stats);
  {
    int n = BSZ * P2 * CCH;
    gn_norm_kernel<<<(n + 255) / 256, 256, 0, stream>>>(pre, stats, gn_w, gn_b,
                                                        shortcut);
  }

  // Q/K/V projections
  gemm(qtok,  wQ16, q_b, nullptr, nullptr, qproj, (int)NTQ, CCH, CCH, 0);
  gemm(kvtok, wK16, k_b, nullptr, nullptr, kproj, (int)NT2, CCH, CCH, 0);
  gemm(kvtok, wV16, v_b, nullptr, nullptr, vproj, (int)NT2, CCH, CCH, 0);

  // flash attention: x1s += attention output
  attention_kernel<<<BSZ * NHEAD * (PQ / 16), 32, 0, stream>>>(qproj, kproj,
                                                               vproj, x1s);

  // FFN with residual
  cvt(x1s, x1s16, (int)(NTQ * CCH));
  gemm(x1s16, wFc1_16, fc1_b, nullptr, nullptr, hidden, (int)NTQ, FFN, CCH, 1);
  gemm(hidden, wFc2_16, fc2_b, x1s, x1s2, nullptr, (int)NTQ, CCH, FFN, 0);

  // upsample 16->96, final conv + shortcut, permute to BCHW
  {
    int n = BSZ * P2 * CCH;
    upsample2_kernel<<<(n + 255) / 256, 256, 0, stream>>>(x1s2, up2);
  }
  gemm(up2, wOut16, out_b, shortcut, ftok, nullptr, (int)NT2, CCH, CCH, 0);
  {
    int n = BSZ * CCH * P2;
    permute_out_kernel<<<(n + 255) / 256, 256, 0, stream>>>(ftok, out);
  }
}
